// InductiveGNN_42760694399653
// MI455X (gfx1250) — compile-verified, matrix operand order checked by reading, not run on hardware
//
#include <hip/hip_runtime.h>
#include <stdint.h>

#define NROWS       200000
#define DIM         128
#define KNEI        16
#define ROWS_PER_WG 128
#define THREADS     256

// padded row strides (bf16 elements): keep 16B alignment, break bank-conflict stride
#define S128 136   // for Kdim=128 weight tiles and activation slices
#define S256 264   // for Wa1 (Kdim=256)

typedef __attribute__((ext_vector_type(16))) __bf16 v16bf;
typedef __attribute__((ext_vector_type(2)))  __bf16 v2bf;
typedef __attribute__((ext_vector_type(8)))  float  v8f;

struct SharedMem {
  uint16_t wbuf[128 * S256];       // 67,584 B : per-stage transposed weights WT[n][k]
  uint16_t hbuf[8][16 * S128];     // 34,816 B : per-wave activation slice (h1/h2/out_bf16/g)
  uint16_t avgbuf[8][16 * S128];   // 34,816 B : per-wave pooled-neighbor slice (bf16)
};                                 // total ~134 KB -> 2 WGs per WGP (320 KB LDS)

// f32 -> bf16 via native fptrunc: gfx1250 has native BF16 VALU, so the backend
// should select v_cvt(_pk)_bf16_f32-class instructions for these casts. If not,
// clang emits the RNE software sequence (same cost as the old manual path).
__device__ __forceinline__ uint16_t f2bf(float f) {
  __bf16 b = (__bf16)f;
  return __builtin_bit_cast(uint16_t, b);
}
__device__ __forceinline__ uint32_t pack2bf(float lo, float hi) {
  v2bf r = {(__bf16)lo, (__bf16)hi};
  return __builtin_bit_cast(uint32_t, r);
}

union Frag {
  uint4    q[2];
  uint32_t u[8];
  v16bf    v;
};

// 16-bit A/B fragment per ISA layout: lane holds K = khalf*8 + [0..8) (VGPR0-3)
// and 16 + khalf*8 + [0..8) (VGPR4-7), pairs packed little-endian -> two b128 reads.
__device__ __forceinline__ void loadFragLDS(Frag& f, const uint16_t* base,
                                            int rowStride, int row, int khalf, int kOff) {
  const uint16_t* p = base + row * rowStride + kOff + khalf * 8;
  f.q[0] = *(const uint4*)p;
  f.q[1] = *(const uint4*)(p + 16);
}

// A fragment straight from a global f32 row (stage 1), converting to bf16 pairs.
__device__ __forceinline__ void loadFragGlobalF32(Frag& f, const float* rowPtr,
                                                  int khalf, int kOff) {
  const float* p = rowPtr + kOff + khalf * 8;
  float4 a0 = *(const float4*)(p + 0);
  float4 a1 = *(const float4*)(p + 4);
  float4 a2 = *(const float4*)(p + 16);
  float4 a3 = *(const float4*)(p + 20);
  f.u[0] = pack2bf(a0.x, a0.y); f.u[1] = pack2bf(a0.z, a0.w);
  f.u[2] = pack2bf(a1.x, a1.y); f.u[3] = pack2bf(a1.z, a1.w);
  f.u[4] = pack2bf(a2.x, a2.y); f.u[5] = pack2bf(a2.z, a2.w);
  f.u[6] = pack2bf(a3.x, a3.y); f.u[7] = pack2bf(a3.z, a3.w);
}

// Cooperative load of W [Kdim][128] f32 -> transposed bf16 WT[n][k] in LDS.
__device__ __forceinline__ void loadWT(const float* __restrict__ W, int Kdim,
                                       uint16_t* wbuf, int stride, int tid) {
  int total = Kdim * 128;
  for (int i = tid; i < total; i += THREADS) {
    int k = i >> 7, n = i & 127;
    wbuf[n * stride + k] = f2bf(W[i]);
  }
}

// One 16xK * Kx128 GEMM stage: 8 N-tiles accumulated over nK K-chunks of 32.
template <typename AF>
__device__ __forceinline__ void gemmTiles(v8f acc[8], AF&& loadA,
                                          const uint16_t* wt, int wStride,
                                          int nK, int idx, int khalf) {
  for (int kc = 0; kc < nK; ++kc) {
    Frag a;
    loadA(a, kc);
#pragma unroll
    for (int t = 0; t < 8; ++t) {
      Frag b;
      loadFragLDS(b, wt, wStride, t * 16 + idx, khalf, kc * 32);
      acc[t] = __builtin_amdgcn_wmma_f32_16x16x32_bf16(
          false, a.v, false, b.v, (short)0, acc[t], false, false);
    }
  }
}

// bias + optional relu, write bf16 tile to the wave's LDS slice (C/D layout:
// lane n = t*16+idx holds M = khalf*8 + r in acc[t][r]).
__device__ __forceinline__ void epilogueToLDS(const v8f acc[8], const float* __restrict__ bias,
                                              uint16_t* buf, int stride, int idx, int khalf,
                                              bool relu) {
#pragma unroll
  for (int t = 0; t < 8; ++t) {
    int n = t * 16 + idx;
    float bv = bias[n];
#pragma unroll
    for (int r = 0; r < 8; ++r) {
      int m = khalf * 8 + r;
      float v = acc[t][r] + bv;
      if (relu) v = fmaxf(v, 0.0f);
      buf[m * stride + n] = f2bf(v);
    }
  }
}

__global__ __launch_bounds__(THREADS, 1)
void InductiveGNN_fused_wmma(const float* __restrict__ x,
                             const float* __restrict__ nf,
                             const unsigned char* __restrict__ mask,
                             const float* __restrict__ W0, const float* __restrict__ b0,
                             const float* __restrict__ W1, const float* __restrict__ b1,
                             const float* __restrict__ Wf, const float* __restrict__ bf_,
                             const float* __restrict__ Wa1, const float* __restrict__ ba1,
                             const float* __restrict__ Wa2, const float* __restrict__ ba2,
                             float* __restrict__ out) {
  __shared__ SharedMem smem;

  const int tid   = threadIdx.x;
  const int wave  = tid >> 5;
  const int lane  = tid & 31;
  const int idx   = lane & 15;
  const int khalf = lane >> 4;
  const int rowBase = blockIdx.x * ROWS_PER_WG + wave * 16;

  uint16_t* hbuf = smem.hbuf[wave];
  uint16_t* avgb = smem.avgbuf[wave];

  // ---- Prologue: mean-pool 16 neighbors for this wave's 16 rows (the big HBM stream).
  // Lane = float4 column chunk (32 chunks cover 128 cols); fully coalesced 512B lines.
  {
    const float inv = 1.0f / (float)KNEI;
#pragma unroll 4
    for (int r = 0; r < 16; ++r) {
      int row  = rowBase + r;
      int crow = row < NROWS ? row : NROWS - 1;
      // prefetch the stream two rows ahead (8KB/row; 128B-line coverage via 2 issues)
      if (r + 2 < 16) {
        int prow = rowBase + r + 2;
        prow = prow < NROWS ? prow : NROWS - 1;
        const char* pf = (const char*)(nf + (size_t)prow * (KNEI * DIM));
        __builtin_prefetch(pf + lane * 128, 0, 0);
        __builtin_prefetch(pf + 4096 + lane * 128, 0, 0);
      }
      const float4* p = (const float4*)(nf + (size_t)crow * (KNEI * DIM)) + lane;
      float4 s = {0.f, 0.f, 0.f, 0.f};
#pragma unroll
      for (int j = 0; j < KNEI; ++j) {
        float4 v = p[j * (DIM / 4)];
        s.x += v.x; s.y += v.y; s.z += v.z; s.w += v.w;
      }
      uint32_t* dst = (uint32_t*)(avgb + r * S128 + lane * 4);
      dst[0] = pack2bf(s.x * inv, s.y * inv);
      dst[1] = pack2bf(s.z * inv, s.w * inv);
    }
  }

  const int arow = (rowBase + idx) < NROWS ? (rowBase + idx) : (NROWS - 1);
  const float* xrow = x + (size_t)arow * DIM;
  const v8f vzero = {0.f, 0.f, 0.f, 0.f, 0.f, 0.f, 0.f, 0.f};

  // ---- Stage 1: h1 = relu(x @ W0 + b0) ; A straight from global f32.
  loadWT(W0, 128, smem.wbuf, S128, tid);
  __syncthreads();
  v8f acc[8];
#pragma unroll
  for (int t = 0; t < 8; ++t) acc[t] = vzero;
  gemmTiles(acc, [&](Frag& a, int kc) { loadFragGlobalF32(a, xrow, khalf, kc * 32); },
            smem.wbuf, S128, 4, idx, khalf);
  epilogueToLDS(acc, b0, hbuf, S128, idx, khalf, true);
  __syncthreads();

  // ---- Stage 2: h2 = relu(h1 @ W1 + b1) ; per-wave slice, in-order LDS allows reuse.
  loadWT(W1, 128, smem.wbuf, S128, tid);
  __syncthreads();
#pragma unroll
  for (int t = 0; t < 8; ++t) acc[t] = vzero;
  gemmTiles(acc, [&](Frag& a, int kc) { loadFragLDS(a, hbuf, S128, idx, khalf, kc * 32); },
            smem.wbuf, S128, 4, idx, khalf);
  epilogueToLDS(acc, b1, hbuf, S128, idx, khalf, true);
  __syncthreads();

  // ---- Stage 3: out = h2 @ Wf + bf ; keep f32 in regs, mirror bf16 copy to hbuf.
  loadWT(Wf, 128, smem.wbuf, S128, tid);
  __syncthreads();
  v8f outAcc[8];
#pragma unroll
  for (int t = 0; t < 8; ++t) outAcc[t] = vzero;
  gemmTiles(outAcc, [&](Frag& a, int kc) { loadFragLDS(a, hbuf, S128, idx, khalf, kc * 32); },
            smem.wbuf, S128, 4, idx, khalf);
#pragma unroll
  for (int t = 0; t < 8; ++t) {
    int n = t * 16 + idx;
    float bv = bf_[n];
#pragma unroll
    for (int r = 0; r < 8; ++r) {
      int m = khalf * 8 + r;
      outAcc[t][r] += bv;
      hbuf[m * S128 + n] = f2bf(outAcc[t][r]);
    }
  }
  __syncthreads();

  // ---- Stage 4: g = relu([out | avg] @ Wa1 + ba1), K = 256 (8 chunks).
  loadWT(Wa1, 256, smem.wbuf, S256, tid);
  __syncthreads();
#pragma unroll
  for (int t = 0; t < 8; ++t) acc[t] = vzero;
  gemmTiles(acc,
            [&](Frag& a, int kc) {
              if (kc < 4) loadFragLDS(a, hbuf, S128, idx, khalf, kc * 32);
              else        loadFragLDS(a, avgb, S128, idx, khalf, (kc - 4) * 32);
            },
            smem.wbuf, S256, 8, idx, khalf);
  epilogueToLDS(acc, ba1, hbuf, S128, idx, khalf, true);   // reads done; in-order LDS
  __syncthreads();

  // ---- Stage 5: enhanced = g @ Wa2 + ba2 ; final select + store.
  loadWT(Wa2, 128, smem.wbuf, S128, tid);
  __syncthreads();
  v8f eacc[8];
#pragma unroll
  for (int t = 0; t < 8; ++t) eacc[t] = vzero;
  gemmTiles(eacc, [&](Frag& a, int kc) { loadFragLDS(a, hbuf, S128, idx, khalf, kc * 32); },
            smem.wbuf, S128, 4, idx, khalf);

  float bav[8];
#pragma unroll
  for (int t = 0; t < 8; ++t) bav[t] = ba2[t * 16 + idx];

#pragma unroll
  for (int r = 0; r < 8; ++r) {
    int m    = khalf * 8 + r;
    int row  = rowBase + m;
    int crow = row < NROWS ? row : NROWS - 1;
    float mk = mask[crow] ? 0.3f : 0.0f;
    bool ok  = row < NROWS;
#pragma unroll
    for (int t = 0; t < 8; ++t) {
      int n = t * 16 + idx;
      float res = outAcc[t][r] + mk * (eacc[t][r] + bav[t]);
      if (ok) out[(size_t)row * DIM + n] = res;
    }
  }
}

extern "C" void kernel_launch(void* const* d_in, const int* in_sizes, int n_in,
                              void* d_out, int out_size, void* d_ws, size_t ws_size,
                              hipStream_t stream) {
  (void)in_sizes; (void)n_in; (void)out_size; (void)d_ws; (void)ws_size;
  const float*         x    = (const float*)d_in[0];
  const float*         nf   = (const float*)d_in[1];
  const unsigned char* mask = (const unsigned char*)d_in[2];   // jnp bool -> 1 byte
  const float* W0  = (const float*)d_in[3];
  const float* b0  = (const float*)d_in[4];
  const float* W1  = (const float*)d_in[5];
  const float* b1  = (const float*)d_in[6];
  const float* Wf  = (const float*)d_in[7];
  const float* bf_ = (const float*)d_in[8];
  const float* Wa1 = (const float*)d_in[9];
  const float* ba1 = (const float*)d_in[10];
  const float* Wa2 = (const float*)d_in[11];
  const float* ba2 = (const float*)d_in[12];
  float* out = (float*)d_out;

  dim3 grid((NROWS + ROWS_PER_WG - 1) / ROWS_PER_WG);
  hipLaunchKernelGGL(InductiveGNN_fused_wmma, grid, dim3(THREADS), 0, stream,
                     x, nf, mask, W0, b0, W1, b1, Wf, bf_, Wa1, ba1, Wa2, ba2, out);
}
